// MultiHeadAttention_69320772157976
// MI455X (gfx1250) — matrix operand order repaired and checked by reading, not compiled, not run on hardware
//
#include <hip/hip_runtime.h>
#include <stdint.h>

#define B_   4
#define T_   2048
#define D_   1024
#define H_   16
#define DK_  64
#define MTOT (B_*T_)          // 8192
#define N3   (3*D_)           // 3072
#define LDSW 40               // padded LDS row stride (elements) = 80 bytes
#define KT_  (D_/32)          // 32 K-tiles
#define KROW 72               // padded K-chunk LDS row stride (elements) = 144 bytes

typedef __attribute__((ext_vector_type(16))) __bf16 v16bf;
typedef __attribute__((ext_vector_type(8)))  __bf16 v8bf;
typedef __attribute__((ext_vector_type(8)))  float  v8f;

union ABf { v16bf v; v8bf h[2]; };

__device__ __forceinline__ v8f wmma_bf16(v16bf a, v16bf b, v8f c) {
  return __builtin_amdgcn_wmma_f32_16x16x32_bf16(false, a, false, b, (short)0, c, false, false);
}

// A fragment (16x32 bf16, row-major source, row stride ld elements).
// lanes 0-15: row=lane, K in {0..7,16..23}; lanes 16-31: row=lane-16, K in {8..15,24..31}
__device__ __forceinline__ v16bf load_frag_a(const __bf16* tile, size_t ld) {
  const int lane = threadIdx.x & 31;
  const size_t row = (size_t)(lane & 15);
  const int kb = (lane >> 4) << 3;                 // 0 or 8
  ABf f;
  f.h[0] = *reinterpret_cast<const v8bf*>(tile + row * ld + kb);
  f.h[1] = *reinterpret_cast<const v8bf*>(tile + row * ld + 16 + kb);
  return f.v;
}

// B fragment (32x16 bf16). B(k,n) = src[n*ld + k].
// lanes 0-15: n=lane, K in {0..7,16..23} ; lanes 16-31: n=lane-16, K in {8..15,24..31}
__device__ __forceinline__ v16bf load_frag_b(const __bf16* src, size_t ld) {
  const int lane = threadIdx.x & 31;
  const size_t n = (size_t)(lane & 15);
  const int ko = (lane >> 4) << 3;                 // 0 or 8
  ABf f;
  f.h[0] = *reinterpret_cast<const v8bf*>(src + n * ld + ko);
  f.h[1] = *reinterpret_cast<const v8bf*>(src + n * ld + ko + 16);
  return f.v;
}

// Stage a 128-row x 32-element (64B) bf16 tile (global row stride D_) into LDS
// (row stride LDSW elements) with async DMA; 2 x b128 per thread, 256 threads.
__device__ __forceinline__ void stage_async_128x32(const __bf16* g, uint32_t lds, int tid) {
#pragma unroll
  for (int j = 0; j < 2; ++j) {
    const int c   = tid + (j << 8);                // 0..511
    const int row = c >> 2;
    const int col = (c & 3) << 4;                  // byte offset 0/16/32/48
    const uint64_t ga = (uint64_t)(uintptr_t)g + (uint64_t)row * (D_ * 2) + (uint64_t)col;
    const uint32_t la = lds + (uint32_t)(row * (LDSW * 2) + col);
    asm volatile("global_load_async_to_lds_b128 %0, %1, off"
                 :: "v"(la), "v"(ga) : "memory");
  }
}

// Stage K chunk: 32 rows (t_k) x 64 elem (128B payload), LDS row stride 144B.
__device__ __forceinline__ void stage_k_chunk(const __bf16* Kh, int kc, uint32_t lds, int tid) {
  const int row = tid >> 3;                        // 0..31
  const int col = (tid & 7) << 4;                  // 0..112
  const uint64_t ga = (uint64_t)(uintptr_t)Kh + (uint64_t)(kc + row) * (DK_ * 2) + (uint64_t)col;
  const uint32_t la = lds + (uint32_t)(row * (KROW * 2) + col);
  asm volatile("global_load_async_to_lds_b128 %0, %1, off"
               :: "v"(la), "v"(ga) : "memory");
}

// Stage V^T chunk: 64 rows (dk) x 32 elem (64B payload), LDS row stride 80B.
__device__ __forceinline__ void stage_v_chunk(const __bf16* Vh, int kc, uint32_t lds, int tid) {
  const int row = tid >> 2;                        // 0..63
  const int col = (tid & 3) << 4;                  // 0..48
  const uint64_t ga = (uint64_t)(uintptr_t)Vh + (uint64_t)row * (T_ * 2) + (uint64_t)kc * 2 + (uint64_t)col;
  const uint32_t la = lds + (uint32_t)(row * (LDSW * 2) + col);
  asm volatile("global_load_async_to_lds_b128 %0, %1, off"
               :: "v"(la), "v"(ga) : "memory");
}

// ---------------- kernel 0: fp32 -> bf16 packing ----------------
__global__ __launch_bounds__(256) void mha_pack(
    const float* __restrict__ x,
    const float* __restrict__ Wq, const float* __restrict__ Wk,
    const float* __restrict__ Wv, const float* __restrict__ Wo,
    __bf16* __restrict__ xb, __bf16* __restrict__ wqkv, __bf16* __restrict__ wob) {
  const size_t stride = (size_t)gridDim.x * blockDim.x;
  const size_t nx = (size_t)MTOT * D_;
  const size_t nw = (size_t)D_ * D_;
  for (size_t i = (size_t)blockIdx.x * blockDim.x + threadIdx.x; i < nx; i += stride)
    xb[i] = (__bf16)x[i];
  for (size_t i = (size_t)blockIdx.x * blockDim.x + threadIdx.x; i < nw; i += stride) {
    wqkv[i]          = (__bf16)Wq[i];
    wqkv[nw + i]     = (__bf16)Wk[i];
    wqkv[2 * nw + i] = (__bf16)Wv[i];
    wob[i]           = (__bf16)Wo[i];
  }
}

// ---------------- kernel 1: fused QKV projection ----------------
__global__ __launch_bounds__(256) void mha_gemm_qkv(
    const __bf16* __restrict__ xb, const __bf16* __restrict__ wqkv,
    const float* __restrict__ bq, const float* __restrict__ bk, const float* __restrict__ bv,
    __bf16* __restrict__ Q, __bf16* __restrict__ K, __bf16* __restrict__ Vt) {
  __shared__ __align__(16) __bf16 Asm[2][128 * LDSW];
  __shared__ __align__(16) __bf16 Bsm[2][128 * LDSW];
  const int tid  = threadIdx.x;
  const int wave = tid >> 5;
  const int lane = tid & 31;
  const int wm = (wave & 3) * 32;
  const int wn = (wave >> 2) * 64;
  const uint32_t lA[2] = {(uint32_t)(uintptr_t)&Asm[0][0], (uint32_t)(uintptr_t)&Asm[1][0]};
  const uint32_t lB[2] = {(uint32_t)(uintptr_t)&Bsm[0][0], (uint32_t)(uintptr_t)&Bsm[1][0]};
  const __bf16* Ag = xb   + (size_t)(blockIdx.x * 128) * D_;
  const __bf16* Bg = wqkv + (size_t)(blockIdx.y * 128) * D_;

  stage_async_128x32(Ag, lA[0], tid);
  stage_async_128x32(Bg, lB[0], tid);

  v8f acc[8] = {};
  for (int kt = 0; kt < KT_; ++kt) {
    const int cur = kt & 1;
    if (kt + 1 < KT_) {
      const int ko = (kt + 1) * 32;
      stage_async_128x32(Ag + ko, lA[cur ^ 1], tid);
      stage_async_128x32(Bg + ko, lB[cur ^ 1], tid);
      asm volatile("s_wait_asynccnt 0x4" ::: "memory");
    } else {
      asm volatile("s_wait_asynccnt 0x0" ::: "memory");
    }
    __syncthreads();
    const __bf16* At = &Asm[cur][wm * LDSW];
    const __bf16* Bt = &Bsm[cur][wn * LDSW];
    const v16bf a0 = load_frag_a(At, LDSW);
    const v16bf a1 = load_frag_a(At + 16 * LDSW, LDSW);
#pragma unroll
    for (int j = 0; j < 4; ++j) {
      v16bf b = load_frag_b(Bt + (size_t)j * 16 * LDSW, LDSW);
      acc[j]     = wmma_bf16(a0, b, acc[j]);
      acc[4 + j] = wmma_bf16(a1, b, acc[4 + j]);
    }
    __syncthreads();
  }

  const int rowhi = (lane >> 4) << 3;
  const int nl = lane & 15;
#pragma unroll
  for (int half = 0; half < 2; ++half) {
#pragma unroll
    for (int j = 0; j < 4; ++j) {
#pragma unroll
      for (int r = 0; r < 8; ++r) {
        const int m = blockIdx.x * 128 + wm + half * 16 + r + rowhi;
        const int n = blockIdx.y * 128 + wn + j * 16 + nl;
        const int which = n >> 10;       // 0=Q,1=K,2=V
        const int f = n & (D_ - 1);
        const int b = m >> 11, t = m & (T_ - 1);
        const int h = f >> 6, dk = f & (DK_ - 1);
        const float bias = (which == 0) ? bq[f] : (which == 1) ? bk[f] : bv[f];
        const __bf16 val = (__bf16)(acc[half * 4 + j][r] + bias);
        if (which == 0)
          Q[(((size_t)(b * H_ + h) * T_ + t) << 6) + dk] = val;
        else if (which == 1)
          K[(((size_t)(b * H_ + h) * T_ + t) << 6) + dk] = val;
        else
          Vt[((size_t)(b * H_ + h) * DK_ + dk) * T_ + t] = val;
      }
    }
  }
}

// ---------------- kernel 2: causal flash attention ----------------
// Block = 8 waves = 128 q rows; K/V chunks (32 keys) async-staged to LDS,
// shared by all waves, double-buffered. grid = (B*H, T/128), block = 256.
__global__ __launch_bounds__(256) void mha_attn(
    const __bf16* __restrict__ Q, const __bf16* __restrict__ K,
    const __bf16* __restrict__ Vt, __bf16* __restrict__ Ao) {
  __shared__ __align__(16) __bf16 Ksm[2][32 * KROW];
  __shared__ __align__(16) __bf16 Vsm[2][64 * LDSW];
  __shared__ __align__(16) __bf16 pbuf[8][16][32];
  const int tid  = threadIdx.x;
  const int wave = tid >> 5;
  const int lane = tid & 31;
  const int bh = blockIdx.x;
  const int q0b = blockIdx.y * 128;
  const int q0 = q0b + wave * 16;
  const __bf16* Qh = Q + (size_t)bh * T_ * DK_;
  const __bf16* Kh = K + (size_t)bh * T_ * DK_;
  const __bf16* Vh = Vt + (size_t)bh * DK_ * T_;
  const uint32_t lK[2] = {(uint32_t)(uintptr_t)&Ksm[0][0], (uint32_t)(uintptr_t)&Ksm[1][0]};
  const uint32_t lV[2] = {(uint32_t)(uintptr_t)&Vsm[0][0], (uint32_t)(uintptr_t)&Vsm[1][0]};

  const v16bf qf0 = load_frag_a(Qh + (size_t)q0 * DK_ + 0, DK_);
  const v16bf qf1 = load_frag_a(Qh + (size_t)q0 * DK_ + 32, DK_);

  const int rowhi = (lane >> 4) << 3;
  const int nl = lane & 15;

  float m_run[8], l_run[8];
  v8f o[4] = {};
#pragma unroll
  for (int r = 0; r < 8; ++r) { m_run[r] = -1e30f; l_run[r] = 0.f; }

  const int nchunks = (q0b + 128) >> 5;            // block-wide causal chunk count
  stage_k_chunk(Kh, 0, lK[0], tid);
  stage_v_chunk(Vh, 0, lV[0], tid);

  for (int c = 0; c < nchunks; ++c) {
    const int kc = c << 5;
    const int cur = c & 1;
    if (c + 1 < nchunks) {
      stage_k_chunk(Kh, kc + 32, lK[cur ^ 1], tid);
      stage_v_chunk(Vh, kc + 32, lV[cur ^ 1], tid);
      asm volatile("s_wait_asynccnt 0x2" ::: "memory");
    } else {
      asm volatile("s_wait_asynccnt 0x0" ::: "memory");
    }
    __syncthreads();
    if (kc < q0 + 16) {                            // wave-uniform causal skip
      v8f s[2] = {};
#pragma unroll
      for (int half = 0; half < 2; ++half) {
        v16bf kb0 = load_frag_b(&Ksm[cur][(half * 16) * KROW], KROW);
        v16bf kb1 = load_frag_b(&Ksm[cur][(half * 16) * KROW + 32], KROW);
        s[half] = wmma_bf16(qf0, kb0, s[half]);
        s[half] = wmma_bf16(qf1, kb1, s[half]);
      }
#pragma unroll
      for (int r = 0; r < 8; ++r) {
        const int qrow = q0 + r + rowhi;
        const int tk0 = kc + nl, tk1 = kc + 16 + nl;
        float v0 = (tk0 <= qrow) ? s[0][r] * 0.125f : -1e30f;
        float v1 = (tk1 <= qrow) ? s[1][r] * 0.125f : -1e30f;
        float mx = fmaxf(v0, v1);
#pragma unroll
        for (int off = 1; off < 16; off <<= 1) mx = fmaxf(mx, __shfl_xor(mx, off, 32));
        const float mnew = fmaxf(m_run[r], mx);
        const float alpha = __expf(m_run[r] - mnew);
        const float p0 = __expf(v0 - mnew);
        const float p1 = __expf(v1 - mnew);
        float rs = p0 + p1;
#pragma unroll
        for (int off = 1; off < 16; off <<= 1) rs += __shfl_xor(rs, off, 32);
        l_run[r] = l_run[r] * alpha + rs;
        m_run[r] = mnew;
#pragma unroll
        for (int j = 0; j < 4; ++j) o[j][r] *= alpha;
        pbuf[wave][r + rowhi][nl]      = (__bf16)p0; // wave-private P tile
        pbuf[wave][r + rowhi][16 + nl] = (__bf16)p1;
      }
      const v16bf pf = load_frag_a(&pbuf[wave][0][0], 32);
#pragma unroll
      for (int j = 0; j < 4; ++j) {
        v16bf vf = load_frag_b(&Vsm[cur][(j * 16) * LDSW], LDSW);
        o[j] = wmma_bf16(pf, vf, o[j]);
      }
    }
    __syncthreads();
  }

  const int b = bh >> 4, h = bh & (H_ - 1);
#pragma unroll
  for (int r = 0; r < 8; ++r) {
    const float inv = 1.0f / l_run[r];
    const int t = q0 + r + rowhi;
#pragma unroll
    for (int j = 0; j < 4; ++j) {
      const int d = j * 16 + nl;
      Ao[(size_t)(b * T_ + t) * D_ + h * DK_ + d] = (__bf16)(o[j][r] * inv);
    }
  }
}

// ---------------- kernel 3: output projection ----------------
__global__ __launch_bounds__(256) void mha_gemm_out(
    const __bf16* __restrict__ Ab, const __bf16* __restrict__ Wob,
    const float* __restrict__ bo, float* __restrict__ out) {
  __shared__ __align__(16) __bf16 Asm[2][128 * LDSW];
  __shared__ __align__(16) __bf16 Bsm[2][128 * LDSW];
  const int tid  = threadIdx.x;
  const int wave = tid >> 5;
  const int lane = tid & 31;
  const int wm = (wave & 3) * 32;
  const int wn = (wave >> 2) * 64;
  const uint32_t lA[2] = {(uint32_t)(uintptr_t)&Asm[0][0], (uint32_t)(uintptr_t)&Asm[1][0]};
  const uint32_t lB[2] = {(uint32_t)(uintptr_t)&Bsm[0][0], (uint32_t)(uintptr_t)&Bsm[1][0]};
  const __bf16* Ag = Ab  + (size_t)(blockIdx.x * 128) * D_;
  const __bf16* Bg = Wob + (size_t)(blockIdx.y * 128) * D_;

  stage_async_128x32(Ag, lA[0], tid);
  stage_async_128x32(Bg, lB[0], tid);

  v8f acc[8] = {};
  for (int kt = 0; kt < KT_; ++kt) {
    const int cur = kt & 1;
    if (kt + 1 < KT_) {
      const int ko = (kt + 1) * 32;
      stage_async_128x32(Ag + ko, lA[cur ^ 1], tid);
      stage_async_128x32(Bg + ko, lB[cur ^ 1], tid);
      asm volatile("s_wait_asynccnt 0x4" ::: "memory");
    } else {
      asm volatile("s_wait_asynccnt 0x0" ::: "memory");
    }
    __syncthreads();
    const __bf16* At = &Asm[cur][wm * LDSW];
    const __bf16* Bt = &Bsm[cur][wn * LDSW];
    const v16bf a0 = load_frag_a(At, LDSW);
    const v16bf a1 = load_frag_a(At + 16 * LDSW, LDSW);
#pragma unroll
    for (int j = 0; j < 4; ++j) {
      v16bf b = load_frag_b(Bt + (size_t)j * 16 * LDSW, LDSW);
      acc[j]     = wmma_bf16(a0, b, acc[j]);
      acc[4 + j] = wmma_bf16(a1, b, acc[4 + j]);
    }
    __syncthreads();
  }

  const int rowhi = (lane >> 4) << 3;
  const int nl = lane & 15;
#pragma unroll
  for (int half = 0; half < 2; ++half) {
#pragma unroll
    for (int j = 0; j < 4; ++j) {
#pragma unroll
      for (int r = 0; r < 8; ++r) {
        const int m = blockIdx.x * 128 + wm + half * 16 + r + rowhi;
        const int n = blockIdx.y * 128 + wn + j * 16 + nl;
        out[(size_t)m * D_ + n] = acc[half * 4 + j][r] + bo[n];
      }
    }
  }
}

extern "C" void kernel_launch(void* const* d_in, const int* in_sizes, int n_in,
                              void* d_out, int out_size, void* d_ws, size_t ws_size,
                              hipStream_t stream) {
  const float* x  = (const float*)d_in[0];
  // d_in[1] = mask (causality computed analytically)
  const float* Wq = (const float*)d_in[2];
  const float* Wk = (const float*)d_in[3];
  const float* Wv = (const float*)d_in[4];
  const float* Wo = (const float*)d_in[5];
  const float* bq = (const float*)d_in[6];
  const float* bk = (const float*)d_in[7];
  const float* bv = (const float*)d_in[8];
  const float* bo = (const float*)d_in[9];
  float* out = (float*)d_out;

  size_t off = 0;
  auto carve = [&](size_t bytes) {
    void* p = (char*)d_ws + off;
    off += (bytes + 255) & ~(size_t)255;
    return p;
  };
  __bf16* xb   = (__bf16*)carve((size_t)MTOT * D_ * 2);
  __bf16* wqkv = (__bf16*)carve((size_t)N3 * D_ * 2);
  __bf16* wob  = (__bf16*)carve((size_t)D_ * D_ * 2);
  __bf16* Qb   = (__bf16*)carve((size_t)MTOT * D_ * 2);
  __bf16* Kb   = (__bf16*)carve((size_t)MTOT * D_ * 2);
  __bf16* Vtb  = (__bf16*)carve((size_t)MTOT * D_ * 2);
  __bf16* Aob  = (__bf16*)carve((size_t)MTOT * D_ * 2);

  hipLaunchKernelGGL(mha_pack, dim3(2048), dim3(256), 0, stream,
                     x, Wq, Wk, Wv, Wo, xb, wqkv, wob);
  hipLaunchKernelGGL(mha_gemm_qkv, dim3(MTOT / 128, N3 / 128), dim3(256), 0, stream,
                     xb, wqkv, bq, bk, bv, Qb, Kb, Vtb);
  hipLaunchKernelGGL(mha_attn, dim3(B_ * H_, T_ / 128), dim3(256), 0, stream,
                     Qb, Kb, Vtb, Aob);
  hipLaunchKernelGGL(mha_gemm_out, dim3(MTOT / 128, D_ / 128), dim3(256), 0, stream,
                     Aob, wob, bo, out);
  (void)in_sizes; (void)n_in; (void)out_size; (void)ws_size;
}
